// MeshNodeBlock_57071525429418
// MI455X (gfx1250) — compile-verified, matrix operand order checked by reading, not run on hardware
//
#include <hip/hip_runtime.h>
#include <hip/hip_bf16.h>

#define D    512
#define DIN  1024

typedef __attribute__((ext_vector_type(16))) __bf16 v16bf;
typedef __attribute__((ext_vector_type(8)))  float  v8f;

union Frag {
    v16bf v;
    uint4 q[2];
};

__device__ __forceinline__ v8f wmma_bf16(const Frag& a, const Frag& b, v8f c) {
    return __builtin_amdgcn_wmma_f32_16x16x32_bf16(
        /*neg_a=*/false, a.v, /*neg_b=*/false, b.v,
        /*c_mod=*/(short)0, c, /*reuse_a=*/false, /*reuse_b=*/false);
}

__device__ __forceinline__ void load_frag(Frag& f, const __bf16* p, int kb) {
    f.q[0] = *(const uint4*)(p + kb);
    f.q[1] = *(const uint4*)(p + kb + 16);
}

// ---------------------------------------------------------------------------
// Kernel 1: copy efeat -> out (passthrough) and scatter-add into agg (fp32).
// One thread per float4 chunk of an edge row (128 chunks / row).
// ---------------------------------------------------------------------------
__global__ __launch_bounds__(256) void k_scatter_copy(
    const float4* __restrict__ efeat4, const int* __restrict__ dst,
    float4* __restrict__ out_e4, float* __restrict__ agg, int nEdges)
{
    long tid = (long)blockIdx.x * blockDim.x + threadIdx.x;
    int e = (int)(tid >> 7);
    int c = (int)(tid & 127);
    if (e >= nEdges) return;
    float4 v = efeat4[(size_t)e * 128 + c];
    out_e4[(size_t)e * 128 + c] = v;
    float* p = agg + (size_t)dst[e] * D + c * 4;
    unsafeAtomicAdd(p + 0, v.x);
    unsafeAtomicAdd(p + 1, v.y);
    unsafeAtomicAdd(p + 2, v.z);
    unsafeAtomicAdd(p + 3, v.w);
}

// ---------------------------------------------------------------------------
// Kernel 2: transpose-convert weights to bf16, [N][K] layout (K contiguous).
// ---------------------------------------------------------------------------
__global__ __launch_bounds__(256) void k_convert_w(
    const float* __restrict__ w1, const float* __restrict__ w2,
    __bf16* __restrict__ w1t, __bf16* __restrict__ w2t)
{
    int i = blockIdx.x * 256 + threadIdx.x;
    if (i < DIN * D) {
        int n = i >> 10;            // output row  (N)
        int k = i & (DIN - 1);      // output col  (K)
        w1t[i] = (__bf16)w1[(size_t)k * D + n];
    } else {
        int j = i - DIN * D;
        if (j < D * D) {
            int n = j >> 9;
            int k = j & (D - 1);
            w2t[j] = (__bf16)w2[(size_t)k * D + n];
        }
    }
}

// ---------------------------------------------------------------------------
// Kernel 3: cat = bf16([agg | nfeat]), [nNodes][1024]. Thread per float4.
// ---------------------------------------------------------------------------
__global__ __launch_bounds__(256) void k_build_cat(
    const float4* __restrict__ agg4, const float4* __restrict__ nfeat4,
    __bf16* __restrict__ cat, int nNodes)
{
    long tid = (long)blockIdx.x * blockDim.x + threadIdx.x;
    int n  = (int)(tid >> 8);
    int kq = (int)(tid & 255);
    if (n >= nNodes) return;
    float4 v = (kq < 128) ? agg4[(size_t)n * 128 + kq]
                          : nfeat4[(size_t)n * 128 + (kq - 128)];
    union { __bf16 h[4]; uint2 u; } pk;
    pk.h[0] = (__bf16)v.x;
    pk.h[1] = (__bf16)v.y;
    pk.h[2] = (__bf16)v.z;
    pk.h[3] = (__bf16)v.w;
    *(uint2*)(cat + (size_t)n * DIN + kq * 4) = pk.u;
}

// ---------------------------------------------------------------------------
// Kernel 4: H = bf16(silu(cat @ w1 + b1)).  M=nNodes, K=1024, N=512.
// Block: 8 waves; block covers 32 rows; wave w covers cols [w*64, w*64+64).
// Wave tile: 2 M-tiles x 4 N-tiles of 16x16 (64 accum VGPRs).
// Software-pipelined: two K-steps per iteration, double-buffered fragments.
// ---------------------------------------------------------------------------
struct Frags1 { Frag a0, a1, f0, f1, f2, f3; };

__device__ __forceinline__ void load_frags1(
    Frags1& F, const __bf16* a0p, const __bf16* a1p,
    const __bf16* bp0, const __bf16* bp1, const __bf16* bp2, const __bf16* bp3,
    int kb)
{
    load_frag(F.a0, a0p, kb);
    load_frag(F.a1, a1p, kb);
    load_frag(F.f0, bp0, kb);
    load_frag(F.f1, bp1, kb);
    load_frag(F.f2, bp2, kb);
    load_frag(F.f3, bp3, kb);
}

__device__ __forceinline__ void wmma_frags1(v8f acc[2][4], const Frags1& F) {
    acc[0][0] = wmma_bf16(F.a0, F.f0, acc[0][0]);
    acc[1][0] = wmma_bf16(F.a1, F.f0, acc[1][0]);
    acc[0][1] = wmma_bf16(F.a0, F.f1, acc[0][1]);
    acc[1][1] = wmma_bf16(F.a1, F.f1, acc[1][1]);
    acc[0][2] = wmma_bf16(F.a0, F.f2, acc[0][2]);
    acc[1][2] = wmma_bf16(F.a1, F.f2, acc[1][2]);
    acc[0][3] = wmma_bf16(F.a0, F.f3, acc[0][3]);
    acc[1][3] = wmma_bf16(F.a1, F.f3, acc[1][3]);
}

__global__ __launch_bounds__(256) void k_gemm1_silu(
    const __bf16* __restrict__ A,   // [M][1024] bf16 (cat)
    const __bf16* __restrict__ Bt,  // [512][1024] bf16 (w1^T)
    const float*  __restrict__ bias1,
    __bf16* __restrict__ H, int M)  // [M][512] bf16
{
    int wave = threadIdx.x >> 5;
    int lane = threadIdx.x & 31;
    int m0 = blockIdx.x * 32;
    int n0 = wave * 64;
    int lm = lane & 15;
    int kh = (lane >> 4) << 3;      // 0 or 8: K-half offset per ISA A/B layout

    int mA0 = m0 + lm;      if (mA0 > M - 1) mA0 = M - 1;
    int mA1 = m0 + 16 + lm; if (mA1 > M - 1) mA1 = M - 1;
    const __bf16* a0p = A  + (size_t)mA0 * DIN + kh;
    const __bf16* a1p = A  + (size_t)mA1 * DIN + kh;
    const __bf16* bp0 = Bt + (size_t)(n0 +  0 + lm) * DIN + kh;
    const __bf16* bp1 = Bt + (size_t)(n0 + 16 + lm) * DIN + kh;
    const __bf16* bp2 = Bt + (size_t)(n0 + 32 + lm) * DIN + kh;
    const __bf16* bp3 = Bt + (size_t)(n0 + 48 + lm) * DIN + kh;

    v8f acc[2][4] = {};
    Frags1 buf0, buf1;

    load_frags1(buf0, a0p, a1p, bp0, bp1, bp2, bp3, 0);
    int kb = 0;
    #pragma unroll 1
    for (; kb < DIN - 64; kb += 64) {
        load_frags1(buf1, a0p, a1p, bp0, bp1, bp2, bp3, kb + 32);
        wmma_frags1(acc, buf0);
        load_frags1(buf0, a0p, a1p, bp0, bp1, bp2, bp3, kb + 64);
        wmma_frags1(acc, buf1);
    }
    load_frags1(buf1, a0p, a1p, bp0, bp1, bp2, bp3, kb + 32);
    wmma_frags1(acc, buf0);
    wmma_frags1(acc, buf1);

    int mloc = (lane >> 4) << 3;    // D layout: row = vgpr + 8*(lane>=16)
    #pragma unroll
    for (int t = 0; t < 4; ++t) {
        int n = n0 + t * 16 + lm;
        float bb = bias1[n];
        #pragma unroll
        for (int mt = 0; mt < 2; ++mt) {
            #pragma unroll
            for (int r = 0; r < 8; ++r) {
                int m = m0 + mt * 16 + mloc + r;
                if (m < M) {
                    float x = acc[mt][t][r] + bb;
                    float y = x / (1.0f + __expf(-x));   // SiLU
                    H[(size_t)m * D + n] = (__bf16)y;
                }
            }
        }
    }
}

// ---------------------------------------------------------------------------
// Kernel 5: out = LayerNorm(H @ w2 + b2)*gamma + beta + nfeat.
// Block: 8 waves covering 16 full rows (each wave 64 cols) so the LN
// reduction over N=512 closes inside the workgroup (shuffle + LDS atomics).
// Software-pipelined K loop like kernel 4.
// ---------------------------------------------------------------------------
struct Frags2 { Frag a, f0, f1, f2, f3; };

__device__ __forceinline__ void load_frags2(
    Frags2& F, const __bf16* ap,
    const __bf16* bp0, const __bf16* bp1, const __bf16* bp2, const __bf16* bp3,
    int kb)
{
    load_frag(F.a, ap, kb);
    load_frag(F.f0, bp0, kb);
    load_frag(F.f1, bp1, kb);
    load_frag(F.f2, bp2, kb);
    load_frag(F.f3, bp3, kb);
}

__device__ __forceinline__ void wmma_frags2(v8f acc[4], const Frags2& F) {
    acc[0] = wmma_bf16(F.a, F.f0, acc[0]);
    acc[1] = wmma_bf16(F.a, F.f1, acc[1]);
    acc[2] = wmma_bf16(F.a, F.f2, acc[2]);
    acc[3] = wmma_bf16(F.a, F.f3, acc[3]);
}

__global__ __launch_bounds__(256) void k_gemm2_ln(
    const __bf16* __restrict__ Hm,  // [M][512] bf16
    const __bf16* __restrict__ Bt,  // [512][512] bf16 (w2^T)
    const float* __restrict__ bias2, const float* __restrict__ gamma,
    const float* __restrict__ beta, const float* __restrict__ nfeat,
    float* __restrict__ out_n, int M)
{
    __shared__ float s_sum[16];
    __shared__ float s_sq[16];

    int wave = threadIdx.x >> 5;
    int lane = threadIdx.x & 31;
    int m0 = blockIdx.x * 16;
    int n0 = wave * 64;
    int lm = lane & 15;
    int kh = (lane >> 4) << 3;

    int mA = m0 + lm; if (mA > M - 1) mA = M - 1;
    const __bf16* ap  = Hm + (size_t)mA * D + kh;
    const __bf16* bp0 = Bt + (size_t)(n0 +  0 + lm) * D + kh;
    const __bf16* bp1 = Bt + (size_t)(n0 + 16 + lm) * D + kh;
    const __bf16* bp2 = Bt + (size_t)(n0 + 32 + lm) * D + kh;
    const __bf16* bp3 = Bt + (size_t)(n0 + 48 + lm) * D + kh;

    v8f acc[4] = {};
    Frags2 buf0, buf1;

    load_frags2(buf0, ap, bp0, bp1, bp2, bp3, 0);
    int kb = 0;
    #pragma unroll 1
    for (; kb < D - 64; kb += 64) {
        load_frags2(buf1, ap, bp0, bp1, bp2, bp3, kb + 32);
        wmma_frags2(acc, buf0);
        load_frags2(buf0, ap, bp0, bp1, bp2, bp3, kb + 64);
        wmma_frags2(acc, buf1);
    }
    load_frags2(buf1, ap, bp0, bp1, bp2, bp3, kb + 32);
    wmma_frags2(acc, buf0);
    wmma_frags2(acc, buf1);

    if (threadIdx.x < 16) { s_sum[threadIdx.x] = 0.0f; s_sq[threadIdx.x] = 0.0f; }
    __syncthreads();

    // Bias, then per-row partial sums (each lane holds 4 n-values per row).
    float ls[8], lq[8];
    #pragma unroll
    for (int r = 0; r < 8; ++r) { ls[r] = 0.0f; lq[r] = 0.0f; }
    #pragma unroll
    for (int t = 0; t < 4; ++t) {
        float bb = bias2[n0 + t * 16 + lm];
        #pragma unroll
        for (int r = 0; r < 8; ++r) {
            float x = acc[t][r] + bb;
            acc[t][r] = x;
            ls[r] += x;
            lq[r] += x * x;
        }
    }
    // Reduce across the 16 lanes that share the same row set (lane bits 0..3).
    #pragma unroll
    for (int off = 1; off < 16; off <<= 1) {
        #pragma unroll
        for (int r = 0; r < 8; ++r) {
            ls[r] += __shfl_xor(ls[r], off, 32);
            lq[r] += __shfl_xor(lq[r], off, 32);
        }
    }
    int mloc = (lane >> 4) << 3;
    if (lm == 0) {
        #pragma unroll
        for (int r = 0; r < 8; ++r) {
            atomicAdd(&s_sum[mloc + r], ls[r]);
            atomicAdd(&s_sq [mloc + r], lq[r]);
        }
    }
    __syncthreads();

    float mu[8], rs[8];
    #pragma unroll
    for (int r = 0; r < 8; ++r) {
        float s = s_sum[mloc + r] * (1.0f / D);
        float q = s_sq [mloc + r] * (1.0f / D);
        mu[r] = s;
        rs[r] = rsqrtf(q - s * s + 1e-5f);
    }
    #pragma unroll
    for (int t = 0; t < 4; ++t) {
        int n = n0 + t * 16 + lm;
        float g = gamma[n], be = beta[n];
        #pragma unroll
        for (int r = 0; r < 8; ++r) {
            int m = m0 + mloc + r;
            if (m < M) {
                out_n[(size_t)m * D + n] =
                    (acc[t][r] - mu[r]) * rs[r] * g + be + nfeat[(size_t)m * D + n];
            }
        }
    }
}

// ---------------------------------------------------------------------------
extern "C" void kernel_launch(void* const* d_in, const int* in_sizes, int n_in,
                              void* d_out, int out_size, void* d_ws, size_t ws_size,
                              hipStream_t stream) {
    const float* efeat = (const float*)d_in[0];
    const float* nfeat = (const float*)d_in[1];
    const int*   dst   = (const int*)  d_in[2];
    const float* w1    = (const float*)d_in[3];
    const float* b1    = (const float*)d_in[4];
    const float* w2    = (const float*)d_in[5];
    const float* b2    = (const float*)d_in[6];
    const float* gamma = (const float*)d_in[7];
    const float* beta  = (const float*)d_in[8];

    int nEdges = in_sizes[0] / D;
    int nNodes = in_sizes[1] / D;

    float* out_e = (float*)d_out;
    float* out_n = out_e + (size_t)nEdges * D;   // node-output section

    // Workspace carve-up (node-output section doubles as the fp32 agg target;
    // it is fully overwritten by k_gemm2_ln at the end).
    float* agg = out_n;
    char* ws = (char*)d_ws;
    size_t o = 0;
    __bf16* cat = (__bf16*)(ws + o); o += (size_t)nNodes * DIN * sizeof(__bf16); o = (o + 255) & ~(size_t)255;
    __bf16* h   = (__bf16*)(ws + o); o += (size_t)nNodes * D   * sizeof(__bf16); o = (o + 255) & ~(size_t)255;
    __bf16* w1t = (__bf16*)(ws + o); o += (size_t)DIN * D      * sizeof(__bf16); o = (o + 255) & ~(size_t)255;
    __bf16* w2t = (__bf16*)(ws + o);

    hipMemsetAsync(agg, 0, (size_t)nNodes * D * sizeof(float), stream);

    {
        long total = (long)nEdges * 128;
        int blocks = (int)((total + 255) / 256);
        k_scatter_copy<<<blocks, 256, 0, stream>>>(
            (const float4*)efeat, dst, (float4*)out_e, agg, nEdges);
    }
    {
        int elems = DIN * D + D * D;
        k_convert_w<<<(elems + 255) / 256, 256, 0, stream>>>(w1, w2, w1t, w2t);
    }
    {
        long total = (long)nNodes * 256;
        k_build_cat<<<(int)((total + 255) / 256), 256, 0, stream>>>(
            (const float4*)agg, (const float4*)nfeat, cat, nNodes);
    }
    {
        int blocks = (nNodes + 31) / 32;
        k_gemm1_silu<<<blocks, 256, 0, stream>>>(cat, w1t, b1, h, nNodes);
    }
    {
        int blocks = (nNodes + 15) / 16;
        k_gemm2_ln<<<blocks, 256, 0, stream>>>(h, w2t, b2, gamma, beta, nfeat, out_n, nNodes);
    }
}
